// SPVBlock_52518860095931
// MI455X (gfx1250) — matrix operand order repaired and checked by reading, not compiled, not run on hardware
//
#include <hip/hip_runtime.h>
#include <hip/hip_bf16.h>
#include <stdint.h>

// ---------------- problem constants (from reference) ----------------
#define NPOINTS  400000
#define NPTS_TOT 1000000
#define MVOX     100000
#define CINCH    64
#define COUTCH   64
#define HALFCH   32
#define SHV      120                       // S/SCALE = 240/2
#define KEYSPACE (4 * SHV * SHV * SHV)     // 6,912,000
#define BN_EPS   1e-5f

typedef float  v8f   __attribute__((ext_vector_type(8)));
typedef __bf16 v16bf __attribute__((ext_vector_type(16)));

// ---------------- workspace layout (bytes) ----------------
static constexpr size_t OFF_SLOT  = 0;                          // int[KEYSPACE]      27,648,000
static constexpr size_t OFF_CTR   = 27648000;                   // int counter (+pad)
static constexpr size_t OFF_INV   = OFF_CTR  + 256;             // int[N]              1,600,000
static constexpr size_t OFF_CNTV  = OFF_INV  + 1600000;         // float[N]            1,600,000
static constexpr size_t OFF_SUMS  = OFF_CNTV + 1600000;         // float[N*64]       102,400,000
static constexpr size_t OFF_IDENT = OFF_SUMS + 102400000;       // bf16[N*64]         51,200,000
static constexpr size_t OFF_HVOX  = OFF_IDENT + 51200000;       // bf16[N*64]         51,200,000
static constexpr size_t OFF_CNTO  = OFF_HVOX + 51200000;        // float[M]              400,000
static constexpr size_t OFF_FRAGS = OFF_CNTO + 400000;          // bf16[42*512]           43,008
static constexpr size_t WS_NEEDED = OFF_FRAGS + 43008;          // ~236.1 MB total

// ---------------- pre-packed B-fragment directory ----------------
// frag = 16x16(n) x 32(k) tile of a weight matrix, stored lane-major:
// frags[frag*512 + lane*16 + j] per the ISA 32x16 bf16 B layout.
#define FB_WIN 0    // W_in 64x64 : 4 nt x 2 kt
#define FB_W1  8    // W1   64x32 : 2 nt x 2 kt
#define FB_W2  12   // W2   32x32 : 2 nt x 1 kt
#define FB_W3  14   // W3   32x64 : 4 nt x 1 kt
#define FB_WO1 18   // Wo1 128x64 : 4 nt x 4 kt
#define FB_WO2 34   // Wo2  64x64 : 4 nt x 2 kt
#define NFRAGS 42
#define NFRAGS_WO 24                      // WO1 + WO2, contiguous at FB_WO1

typedef int v4i __attribute__((vector_size(16)));
typedef __attribute__((address_space(1))) v4i gv4i_t;   // global (prints as __device__)
typedef __attribute__((address_space(3))) v4i lv4i_t;   // LDS    (prints as __shared__)

__device__ __forceinline__ float lrelu(float x) { return x >= 0.f ? x : 0.1f * x; }

__device__ __forceinline__ v8f wmma_bf16(v16bf a, v16bf b, v8f c) {
  return __builtin_amdgcn_wmma_f32_16x16x32_bf16(false, a, false, b, (short)0, c,
                                                 false, false);
}

// Async Global->LDS 16B copy (CDNA5 GLOBAL_LOAD_ASYNC_TO_LDS_B128, ASYNCcnt).
__device__ __forceinline__ void async_copy16(const __bf16* gsrc, __bf16* ldst) {
#if __has_builtin(__builtin_amdgcn_global_load_async_to_lds_b128)
  __builtin_amdgcn_global_load_async_to_lds_b128(
      (gv4i_t*)const_cast<__bf16*>(gsrc), (lv4i_t*)ldst, 0, 0);
#else
  *(uint4*)ldst = *(const uint4*)gsrc;
#endif
}

__device__ __forceinline__ void wait_async0() {
#if __has_builtin(__builtin_amdgcn_s_wait_asynccnt)
  __builtin_amdgcn_s_wait_asynccnt(0);
#else
  asm volatile("s_wait_asynccnt 0" ::: "memory");
#endif
}

// Pre-packed B fragment: one 32B vector load per lane (global or LDS source).
__device__ __forceinline__ v16bf load_frag(const __bf16* frags, int fragIdx, int lane) {
  return *(const v16bf*)(frags + (fragIdx << 9) + (lane << 4));
}

// A fragment (16x32 bf16, row-major source, 16B-aligned rows).
// ISA layout: lane m = lane&15; K chunks [h8,h8+8) and [16+h8,16+h8+8), h8=8*(lane>=16).
__device__ __forceinline__ v16bf load_a_bf16(const __bf16* row, int kBase, int lane) {
  const int h8 = (lane >> 4) << 3;
  union { v16bf v; uint4 q[2]; } u;
  u.q[0] = *(const uint4*)(row + kBase + h8);
  u.q[1] = *(const uint4*)(row + kBase + 16 + h8);
  return u.v;
}

__device__ __forceinline__ v16bf load_a_f32(const float* row, int kBase, int lane) {
  const int h8 = (lane >> 4) << 3;
  v16bf a;
#pragma unroll
  for (int j = 0; j < 8; ++j) a[j] = (__bf16)row[kBase + h8 + j];
#pragma unroll
  for (int j = 0; j < 8; ++j) a[8 + j] = (__bf16)row[kBase + 16 + h8 + j];
  return a;
}

__device__ __forceinline__ int point_key(const int* coors, int i) {
  int b = coors[i * 4 + 0];
  int x = coors[i * 4 + 1] >> 1;   // // SCALE(=2), coords nonnegative
  int y = coors[i * 4 + 2] >> 1;
  int z = coors[i * 4 + 3] >> 1;
  return ((b * SHV + x) * SHV + y) * SHV + z;
}

// ---------------- kernels ----------------
__global__ void k_init(int* slot, int* counter, float* sums, float* cntv,
                       float* cnto, float* outp) {
  int i = blockIdx.x * blockDim.x + threadIdx.x;
  if (i < NPOINTS * CINCH) sums[i] = 0.f;       // largest range: 25.6M
  if (i < KEYSPACE)        slot[i] = -1;
  if (i < NPOINTS)         cntv[i] = 0.f;
  if (i < MVOX)            cnto[i] = 0.f;
  if (i < MVOX * COUTCH)   outp[i] = 0.f;
  if (i == 0)              *counter = 0;
}

__global__ void k_prep_frags(const float* W_in, const float* W1, const float* W2,
                             const float* W3, const float* Wo1, const float* Wo2,
                             __bf16* frags) {
  int t = blockIdx.x * blockDim.x + threadIdx.x;
  if (t >= NFRAGS * 512) return;
  int frag = t >> 9, rem = t & 511, lane = rem >> 4, j = rem & 15;
  const float* W; int ld, nkt, fb;
  if (frag < FB_W1)       { W = W_in; ld = 64; nkt = 2; fb = FB_WIN; }
  else if (frag < FB_W2)  { W = W1;   ld = 32; nkt = 2; fb = FB_W1;  }
  else if (frag < FB_W3)  { W = W2;   ld = 32; nkt = 1; fb = FB_W2;  }
  else if (frag < FB_WO1) { W = W3;   ld = 64; nkt = 1; fb = FB_W3;  }
  else if (frag < FB_WO2) { W = Wo1;  ld = 64; nkt = 4; fb = FB_WO1; }
  else                    { W = Wo2;  ld = 64; nkt = 2; fb = FB_WO2; }
  int f  = frag - fb;
  int nt = f / nkt, kt = f % nkt;
  int n  = nt * 16 + (lane & 15);
  int k  = kt * 32 + ((lane >> 4) << 3) + (j & 7) + ((j >> 3) << 4);   // B layout: lane half h holds K 8h..8h+7 and 16+8h..16+8h+7
  frags[t] = (__bf16)W[k * ld + n];
}

__global__ void k_mark(const int* coors, int* slot) {
  int i = blockIdx.x * blockDim.x + threadIdx.x;
  if (i >= NPOINTS) return;
  slot[point_key(coors, i)] = -2;
}

__global__ void k_assign(int* slot, int* counter) {
  int k = blockIdx.x * blockDim.x + threadIdx.x;
  if (k >= KEYSPACE) return;
  if (slot[k] == -2) slot[k] = atomicAdd(counter, 1);
}

__global__ void k_scatter(const int* coors, const float* feats, const int* slot,
                          int* inv, float* cntv, float* sums) {
  int t = blockIdx.x * blockDim.x + threadIdx.x;
  if (t >= NPOINTS * 8) return;
  int i = t >> 3, c0 = (t & 7) << 3;
  int id = slot[point_key(coors, i)];
  if (c0 == 0) { inv[i] = id; atomicAdd(&cntv[id], 1.0f); }
  const float* f = feats + (size_t)i * CINCH + c0;
  float*       s = sums  + (size_t)id * CINCH + c0;
#pragma unroll
  for (int k = 0; k < 8; ++k) atomicAdd(&s[k], f[k]);
}

// identity = lrelu(features @ W_in + b_in), stored bf16. One 16-row tile per wave.
__global__ void __launch_bounds__(256)
k_identity(const float* feats, const float* b_in, const __bf16* frags, __bf16* ident) {
  __shared__ __align__(16) __bf16 sm[8][16 * 64];
  int lane = threadIdx.x & 31, wave = threadIdx.x >> 5;
  int tile = blockIdx.x * 8 + wave;
  if (tile >= NPOINTS / 16) return;
  int base = tile * 16;
  int nn = lane & 15, mb = (lane >> 4) << 3;
  const float* row = feats + (size_t)(base + nn) * CINCH;
  v16bf a0 = load_a_f32(row, 0, lane);
  v16bf a1 = load_a_f32(row, 32, lane);
  __bf16* s = sm[wave];
#pragma unroll
  for (int nt = 0; nt < 4; ++nt) {
    v8f c = {};
    c = wmma_bf16(a0, load_frag(frags, FB_WIN + nt * 2 + 0, lane), c);
    c = wmma_bf16(a1, load_frag(frags, FB_WIN + nt * 2 + 1, lane), c);
    int n = nt * 16 + nn;
    float bias = b_in[n];
#pragma unroll
    for (int r = 0; r < 8; ++r) s[(mb + r) * 64 + n] = (__bf16)lrelu(c[r] + bias);
  }
#pragma unroll
  for (int it = 0; it < 4; ++it) {            // coalesced 16B flush of 16x64 tile
    int idx = lane + it * 32, r = idx >> 3, ch = idx & 7;
    *(uint4*)(ident + (size_t)(base + r) * 64 + ch * 8) =
        *(const uint4*)(s + r * 64 + ch * 8);
  }
}

// Per-voxel MLP: down=sums/cnt; lrelu(.@W1+b1); bn1; lrelu(.@W2+b2); bn2; lrelu(.@W3+b3).
__global__ void __launch_bounds__(256)
k_voxel(const float* sums, const float* cntv, const int* counter, const __bf16* frags,
        const float* b1, const float* g1, const float* be1, const float* m1, const float* v1,
        const float* b2, const float* g2, const float* be2, const float* m2, const float* v2,
        const float* b3, __bf16* hvox) {
  __shared__ __align__(16) __bf16 sm[8][16 * 64];
  int lane = threadIdx.x & 31, wave = threadIdx.x >> 5;
  int tile = blockIdx.x * 8 + wave;
  if (tile >= NPOINTS / 16) return;
  int base = tile * 16;
  int U = *counter;
  if (base >= U) return;                       // uniform skip past unique count
  int nn = lane & 15, mb = (lane >> 4) << 3;
  __bf16* s = sm[wave];
#pragma unroll 4
  for (int it = 0; it < 32; ++it) {            // stage 16x64 down tile as bf16
    int idx = lane + it * 32, r = idx >> 6, c = idx & 63;
    float cnt = fmaxf(cntv[base + r], 1.0f);
    s[idx] = (__bf16)(sums[(size_t)(base + r) * 64 + c] / cnt);
  }
  // GEMM1: 16x64 @ 64x32
  {
    const __bf16* arow = s + nn * 64;
    v16bf a0 = load_a_bf16(arow, 0, lane);
    v16bf a1 = load_a_bf16(arow, 32, lane);
    v8f cc[2] = {v8f{}, v8f{}};
    cc[0] = wmma_bf16(a0, load_frag(frags, FB_W1 + 0, lane), cc[0]);
    cc[0] = wmma_bf16(a1, load_frag(frags, FB_W1 + 1, lane), cc[0]);
    cc[1] = wmma_bf16(a0, load_frag(frags, FB_W1 + 2, lane), cc[1]);
    cc[1] = wmma_bf16(a1, load_frag(frags, FB_W1 + 3, lane), cc[1]);
#pragma unroll
    for (int nt = 0; nt < 2; ++nt) {
      int n = nt * 16 + nn;
      float bias = b1[n], sc = g1[n] * rsqrtf(v1[n] + BN_EPS), mu = m1[n], bb = be1[n];
#pragma unroll
      for (int r = 0; r < 8; ++r) {
        float h = lrelu(cc[nt][r] + bias);
        s[(mb + r) * 32 + n] = (__bf16)((h - mu) * sc + bb);
      }
    }
  }
  // GEMM2: 16x32 @ 32x32
  {
    const __bf16* arow = s + nn * 32;
    v16bf a0 = load_a_bf16(arow, 0, lane);
    v8f cc[2] = {v8f{}, v8f{}};
    cc[0] = wmma_bf16(a0, load_frag(frags, FB_W2 + 0, lane), cc[0]);
    cc[1] = wmma_bf16(a0, load_frag(frags, FB_W2 + 1, lane), cc[1]);
#pragma unroll
    for (int nt = 0; nt < 2; ++nt) {
      int n = nt * 16 + nn;
      float bias = b2[n], sc = g2[n] * rsqrtf(v2[n] + BN_EPS), mu = m2[n], bb = be2[n];
#pragma unroll
      for (int r = 0; r < 8; ++r) {
        float h = lrelu(cc[nt][r] + bias);
        s[(mb + r) * 32 + n] = (__bf16)((h - mu) * sc + bb);
      }
    }
  }
  // GEMM3: 16x32 @ 32x64
  {
    const __bf16* arow = s + nn * 32;
    v16bf a0 = load_a_bf16(arow, 0, lane);
#pragma unroll
    for (int nt = 0; nt < 4; ++nt) {
      v8f c = {};
      c = wmma_bf16(a0, load_frag(frags, FB_W3 + nt, lane), c);
      int n = nt * 16 + nn;
      float bias = b3[n];
#pragma unroll
      for (int r = 0; r < 8; ++r) s[(mb + r) * 64 + n] = (__bf16)lrelu(c[r] + bias);
    }
  }
#pragma unroll
  for (int it = 0; it < 4; ++it) {
    int idx = lane + it * 32, r = idx >> 3, ch = idx & 7;
    if (base + r < U)
      *(uint4*)(hvox + (size_t)(base + r) * 64 + ch * 8) =
          *(const uint4*)(s + r * 64 + ch * 8);
  }
}

// Per-point (1M) head: g=[identity[p], hvox[inv[p]]] (128ch) -> lrelu(g@Wo1+bo1)@Wo2+bo2,
// segment-accumulated into d_out by scale_coors_inv.
// Wo1/Wo2 B-fragments (24 KB) staged once per block into LDS via async copy.
__global__ void __launch_bounds__(256)
k_out(const int* coors_inv, const int* inv, const int* scale_inv,
      const __bf16* ident, const __bf16* hvox, const __bf16* frags,
      const float* bo1, const float* bo2, float* outp, float* cnto) {
  __shared__ __align__(16) __bf16 sm[8][16 * 64];
  __shared__ __align__(16) __bf16 wf[NFRAGS_WO * 512];
  __shared__ int vb[8][16];
  int lane = threadIdx.x & 31, wave = threadIdx.x >> 5;
  int tile = blockIdx.x * 8 + wave;

  // Block-cooperative async stage of the 24 contiguous WO fragments (1536 x 16B).
  {
    const __bf16* src = frags + (FB_WO1 << 9);
#pragma unroll
    for (int it = 0; it < 6; ++it) {
      int idx = (int)threadIdx.x + it * 256;
      async_copy16(src + idx * 8, wf + idx * 8);
    }
    wait_async0();
    __syncthreads();          // before tile guard: no divergence at the barrier
  }
  if (tile >= NPTS_TOT / 16) return;

  int base = tile * 16;
  int nn = lane & 15, mb = (lane >> 4) << 3;
  int p = coors_inv[base + nn];
  int q = inv[p];
  const __bf16* rI = ident + (size_t)p * 64;
  const __bf16* rH = hvox + (size_t)q * 64;
  v16bf a0 = load_a_bf16(rI, 0, lane);
  v16bf a1 = load_a_bf16(rI, 32, lane);
  v16bf a2 = load_a_bf16(rH, 0, lane);
  v16bf a3 = load_a_bf16(rH, 32, lane);
  __bf16* s = sm[wave];
#pragma unroll
  for (int nt = 0; nt < 4; ++nt) {
    v8f c = {};
    c = wmma_bf16(a0, load_frag(wf, nt * 4 + 0, lane), c);
    c = wmma_bf16(a1, load_frag(wf, nt * 4 + 1, lane), c);
    c = wmma_bf16(a2, load_frag(wf, nt * 4 + 2, lane), c);
    c = wmma_bf16(a3, load_frag(wf, nt * 4 + 3, lane), c);
    int n = nt * 16 + nn;
    float bias = bo1[n];
#pragma unroll
    for (int r = 0; r < 8; ++r) s[(mb + r) * 64 + n] = (__bf16)lrelu(c[r] + bias);
  }
  if (lane < 16) vb[wave][lane] = scale_inv[base + lane];
  const __bf16* arow = s + nn * 64;
  v16bf t0 = load_a_bf16(arow, 0, lane);
  v16bf t1 = load_a_bf16(arow, 32, lane);
#pragma unroll
  for (int nt = 0; nt < 4; ++nt) {
    v8f d = {};
    d = wmma_bf16(t0, load_frag(wf, 16 + nt * 2 + 0, lane), d);
    d = wmma_bf16(t1, load_frag(wf, 16 + nt * 2 + 1, lane), d);
    int n = nt * 16 + nn;
    float bias = bo2[n];
#pragma unroll
    for (int r = 0; r < 8; ++r) {
      int v = vb[wave][mb + r];
      atomicAdd(&outp[(size_t)v * 64 + n], d[r] + bias);
    }
  }
  if (lane < 16) atomicAdd(&cnto[vb[wave][lane]], 1.0f);
}

__global__ void k_div(float* outp, const float* cnto) {
  int i = blockIdx.x * blockDim.x + threadIdx.x;
  if (i >= MVOX * COUTCH) return;
  outp[i] /= fmaxf(cnto[i >> 6], 1.0f);
}

// ---------------- launch ----------------
extern "C" void kernel_launch(void* const* d_in, const int* in_sizes, int n_in,
                              void* d_out, int out_size, void* d_ws, size_t ws_size,
                              hipStream_t stream) {
  (void)in_sizes; (void)n_in; (void)out_size;
  if (ws_size < WS_NEEDED) return;  // workspace too small; nothing safe to do

  const float* features  = (const float*)d_in[0];
  const int*   coors     = (const int*)d_in[1];
  const int*   coors_inv = (const int*)d_in[2];
  const int*   scale_inv = (const int*)d_in[3];
  const float* W_in = (const float*)d_in[4],  *b_in = (const float*)d_in[5];
  const float* W1 = (const float*)d_in[6],  *b1 = (const float*)d_in[7];
  const float* g1 = (const float*)d_in[8],  *be1 = (const float*)d_in[9];
  const float* m1 = (const float*)d_in[10], *v1 = (const float*)d_in[11];
  const float* W2 = (const float*)d_in[12], *b2 = (const float*)d_in[13];
  const float* g2 = (const float*)d_in[14], *be2 = (const float*)d_in[15];
  const float* m2 = (const float*)d_in[16], *v2 = (const float*)d_in[17];
  const float* W3 = (const float*)d_in[18], *b3 = (const float*)d_in[19];
  const float* Wo1 = (const float*)d_in[20], *bo1 = (const float*)d_in[21];
  const float* Wo2 = (const float*)d_in[22], *bo2 = (const float*)d_in[23];
  float* outp = (float*)d_out;

  char* ws = (char*)d_ws;
  int*    slot    = (int*)(ws + OFF_SLOT);
  int*    counter = (int*)(ws + OFF_CTR);
  int*    inv     = (int*)(ws + OFF_INV);
  float*  cntv    = (float*)(ws + OFF_CNTV);
  float*  sums    = (float*)(ws + OFF_SUMS);
  __bf16* ident   = (__bf16*)(ws + OFF_IDENT);
  __bf16* hvox    = (__bf16*)(ws + OFF_HVOX);
  float*  cnto    = (float*)(ws + OFF_CNTO);
  __bf16* frags   = (__bf16*)(ws + OFF_FRAGS);

  const int T = 256;
  k_init<<<(NPOINTS * CINCH + T - 1) / T, T, 0, stream>>>(slot, counter, sums, cntv,
                                                          cnto, outp);
  k_prep_frags<<<(NFRAGS * 512 + T - 1) / T, T, 0, stream>>>(W_in, W1, W2, W3, Wo1,
                                                             Wo2, frags);
  k_mark<<<(NPOINTS + T - 1) / T, T, 0, stream>>>(coors, slot);
  k_assign<<<(KEYSPACE + T - 1) / T, T, 0, stream>>>(slot, counter);
  k_scatter<<<(NPOINTS * 8 + T - 1) / T, T, 0, stream>>>(coors, features, slot, inv,
                                                         cntv, sums);
  k_identity<<<(NPOINTS / 16 + 7) / 8, T, 0, stream>>>(features, b_in, frags, ident);
  k_voxel<<<(NPOINTS / 16 + 7) / 8, T, 0, stream>>>(sums, cntv, counter, frags,
                                                    b1, g1, be1, m1, v1,
                                                    b2, g2, be2, m2, v2, b3, hvox);
  k_out<<<(NPTS_TOT / 16 + 7) / 8, T, 0, stream>>>(coors_inv, inv, scale_inv, ident,
                                                   hvox, frags, bo1, bo2, outp, cnto);
  k_div<<<(MVOX * COUTCH + T - 1) / T, T, 0, stream>>>(outp, cnto);
}
